// CogKR_27178553049430
// MI455X (gfx1250) — matrix operand order repaired and checked by reading, not compiled, not run on hardware
//
#include <hip/hip_runtime.h>
#include <hip/hip_bf16.h>

// ---------------------------------------------------------------------------
// Types for CDNA5 WMMA (wave32, 16x16x32 bf16 -> f32)
// ---------------------------------------------------------------------------
typedef __bf16 bf16x16 __attribute__((ext_vector_type(16)));
typedef __bf16 bf16x8  __attribute__((ext_vector_type(8)));
typedef float  f32x8   __attribute__((ext_vector_type(8)));

__device__ __forceinline__ unsigned short f2bf(float f) {
    unsigned int u = __float_as_uint(f);
    unsigned int r = u + 0x7fffu + ((u >> 16) & 1u);   // round to nearest even
    return (unsigned short)(r >> 16);
}

__device__ __forceinline__ f32x8 bcast8(float x) {
    f32x8 v;
#pragma unroll
    for (int i = 0; i < 8; ++i) v[i] = x;
    return v;
}

// A fragment: 16x32 bf16 tile, row-major in LDS (stride = elements/row).
// Layout used here (contiguous K, matches B packing): lane<16 -> row=lane, K {k0+0..15};
//             lane>=16 -> row=lane-16, K {k0+16..31}.
__device__ __forceinline__ bf16x16 a_frag(const unsigned short* As, int stride, int k0) {
    const int l  = threadIdx.x & 31;
    const int r  = l & 15;
    const int ko = (l < 16) ? 0 : 16;
    bf16x8 lo = *(const bf16x8*)(As + r * stride + k0 + ko);
    bf16x8 hi = *(const bf16x8*)(As + r * stride + k0 + ko + 8);
    bf16x16 f;
#pragma unroll
    for (int e = 0; e < 8; ++e) { f[e] = lo[e]; f[8 + e] = hi[e]; }
    return f;
}

// B fragment: pre-packed in per-lane order (lane*16 contiguous bf16).
__device__ __forceinline__ bf16x16 b_frag(const unsigned short* Bp) {
    return *(const bf16x16*)(Bp + (threadIdx.x & 31) * 16);
}

__device__ __forceinline__ f32x8 wmma_bf16(bf16x16 a, bf16x16 b, f32x8 c) {
    return __builtin_amdgcn_wmma_f32_16x16x32_bf16(false, a, false, b, (short)0, c, false, false);
}

__device__ __forceinline__ float sigm(float x) { return 1.f / (1.f + __expf(-x)); }

// ---------------------------------------------------------------------------
// Weight packing into B-fragment order.
// Packed tile (kt, nt): 512 bf16; element (lane, e):
//   col n = nt*16 + (lane&15); k = kt*32 + ((lane<16)?0:16) + e
// ---------------------------------------------------------------------------
__global__ __launch_bounds__(256)
void pack_b_kernel(const float* __restrict__ src, unsigned short* __restrict__ dst,
                   int K, int N, int transposed) {
    int idx = blockIdx.x * 256 + threadIdx.x;
    int total = (K >> 5) * (N >> 4) * 512;
    if (idx >= total) return;
    int e = idx & 15, lanei = (idx >> 4) & 31, tile = idx >> 9;
    int ntiles = N >> 4;
    int kt = tile / ntiles, nt = tile - kt * ntiles;
    int c = lanei & 15;
    int kk = ((lanei < 16) ? 0 : 16) + e;
    int k = kt * 32 + kk, n = nt * 16 + c;
    float v = transposed ? src[n * K + k] : src[k * N + n];
    dst[idx] = f2bf(v);
}

// W1 = [Ws (256 rows); Wr (128 rows); Wqr (128 rows)], K=512, N=256
__global__ __launch_bounds__(256)
void pack_w1_kernel(const float* __restrict__ Ws, const float* __restrict__ Wr,
                    const float* __restrict__ Wqr, unsigned short* __restrict__ dst) {
    int idx = blockIdx.x * 256 + threadIdx.x;   // total = 16*16*512 = 131072
    int e = idx & 15, lanei = (idx >> 4) & 31, tile = idx >> 9;
    int kt = tile >> 4, nt = tile & 15;
    int c = lanei & 15;
    int kk = ((lanei < 16) ? 0 : 16) + e;
    int k = kt * 32 + kk, n = nt * 16 + c;
    float v;
    if (k < 256)      v = Ws[k * 256 + n];
    else if (k < 384) v = Wr[(k - 256) * 256 + n];
    else              v = Wqr[(k - 384) * 256 + n];
    dst[idx] = f2bf(v);
}

// qdot[b] = sum_q query_repr[b,q] * rank_W[256+q]
__global__ __launch_bounds__(256)
void qdot_kernel(const float* __restrict__ query_repr, const float* __restrict__ rank_W,
                 float* __restrict__ qdot) {
    __shared__ float red[256];
    int b = blockIdx.x, t = threadIdx.x;
    red[t] = query_repr[b * 256 + t] * rank_W[256 + t];
    __syncthreads();
    for (int s = 128; s > 0; s >>= 1) {
        if (t < s) red[t] += red[t + s];
        __syncthreads();
    }
    if (t == 0) qdot[b] = red[0];
}

// ---------------------------------------------------------------------------
// Stage 1: per-edge attention alpha.  A=[hs|hr|hq] (16x512) @ W1 (512x256),
// fused relu -> dot(w_alpha) -> sigmoid.  One wave per 16 edges.
// ---------------------------------------------------------------------------
__global__ __launch_bounds__(32)
void attn_alpha_kernel(const float* __restrict__ node_hidden,
                       const float* __restrict__ relation_emb,
                       const unsigned short* __restrict__ W1p,
                       const float* __restrict__ bqr,
                       const float* __restrict__ w_alpha,
                       const float* __restrict__ b_alpha,
                       const int* __restrict__ head_node,
                       const int* __restrict__ edge_rel,
                       const int* __restrict__ query_rel,
                       float* __restrict__ alpha_out) {
    __shared__ unsigned short As[16 * 512];
    __shared__ float sAl[16];
    __shared__ int sH[16], sR[16], sQ[16];
    const int lane = threadIdx.x;
    const int e0 = blockIdx.x * 16;
    if (lane < 16) {
        sH[lane] = head_node[e0 + lane];
        sR[lane] = edge_rel[e0 + lane];
        sQ[lane] = query_rel[e0 + lane];
        sAl[lane] = 0.f;
    }
    __syncthreads();
    for (int i = lane; i < 16 * 256; i += 32) { int r = i >> 8, c = i & 255; As[r * 512 + c]        = f2bf(node_hidden[sH[r] * 256 + c]); }
    for (int i = lane; i < 16 * 128; i += 32) { int r = i >> 7, c = i & 127; As[r * 512 + 256 + c]  = f2bf(relation_emb[sR[r] * 128 + c]); }
    for (int i = lane; i < 16 * 128; i += 32) { int r = i >> 7, c = i & 127; As[r * 512 + 384 + c]  = f2bf(relation_emb[sQ[r] * 128 + c]); }
    __syncthreads();
    const int c = lane & 15;
    const int rbase = (lane < 16) ? 0 : 8;
    for (int nt = 0; nt < 16; ++nt) {
        f32x8 acc = bcast8(bqr[nt * 16 + c]);
        for (int kt = 0; kt < 16; ++kt)
            acc = wmma_bf16(a_frag(As, 512, kt * 32), b_frag(W1p + (kt * 16 + nt) * 512), acc);
        float wa = w_alpha[nt * 16 + c];
#pragma unroll
        for (int v = 0; v < 8; ++v) {
            float x = acc[v];
            x = x > 0.f ? x : 0.f;                 // relu
            atomicAdd(&sAl[rbase + v], x * wa);
        }
    }
    __syncthreads();
    if (lane < 16) alpha_out[e0 + lane] = sigm(sAl[lane] + b_alpha[0]);
}

// ---------------------------------------------------------------------------
// Stage 2: msg = alpha * [hr | te], atomic scatter-add into agg[tail_node]
// ---------------------------------------------------------------------------
__global__ __launch_bounds__(256)
void scatter_kernel(const float* __restrict__ relation_emb,
                    const float* __restrict__ entity_emb,
                    const float* __restrict__ alpha,
                    const int* __restrict__ edge_rel,
                    const int* __restrict__ tail_ent,
                    const int* __restrict__ tail_node,
                    float* __restrict__ agg) {
    int idx = blockIdx.x * 256 + threadIdx.x;   // M*256 threads
    int e = idx >> 8, c = idx & 255;
    float a = alpha[e];
    float v = (c < 128) ? relation_emb[edge_rel[e] * 128 + c]
                        : entity_emb[tail_ent[e] * 128 + (c - 128)];
    atomicAdd(&agg[tail_node[e] * 256 + c], v * a);
}

// ---------------------------------------------------------------------------
// Stage 3: fused We2h + GRU gates + LayerNorm over 16 tail nodes per wave.
// ---------------------------------------------------------------------------
__global__ __launch_bounds__(32)
void gru_kernel(const float* __restrict__ entity_emb,
                const float* __restrict__ agg,
                const unsigned short* __restrict__ We2hP, const float* __restrict__ We2h_b,
                const unsigned short* __restrict__ WihP, const unsigned short* __restrict__ WhhP,
                const float* __restrict__ b_ih, const float* __restrict__ b_hh,
                const float* __restrict__ ln_g, const float* __restrict__ ln_b,
                const int* __restrict__ tail_node_ent,
                unsigned short* __restrict__ hnew_bf) {
    __shared__ unsigned short sEnt[16 * 128];
    __shared__ unsigned short sAgg[16 * 256];
    __shared__ unsigned short sHp[16 * 256];
    __shared__ float fHp[16 * 256];
    __shared__ float fHn[16 * 256];
    __shared__ float sSum[16], sSq[16];
    __shared__ int sIdx[16];
    const int lane = threadIdx.x;
    const int r0 = blockIdx.x * 16;
    if (lane < 16) { sIdx[lane] = tail_node_ent[r0 + lane]; sSum[lane] = 0.f; sSq[lane] = 0.f; }
    __syncthreads();
    for (int i = lane; i < 16 * 128; i += 32) { int r = i >> 7, c = i & 127; sEnt[i] = f2bf(entity_emb[sIdx[r] * 128 + c]); }
    for (int i = lane; i < 16 * 256; i += 32) { int r = i >> 8, c = i & 255; sAgg[i] = f2bf(agg[(r0 + r) * 256 + c]); }
    __syncthreads();
    const int c = lane & 15;
    const int rbase = (lane < 16) ? 0 : 8;

    // h_prev = lrelu(ent @ We2h + b)   (K=128)
    for (int nt = 0; nt < 16; ++nt) {
        f32x8 acc = bcast8(We2h_b[nt * 16 + c]);
        for (int kt = 0; kt < 4; ++kt)
            acc = wmma_bf16(a_frag(sEnt, 128, kt * 32), b_frag(We2hP + (kt * 16 + nt) * 512), acc);
#pragma unroll
        for (int v = 0; v < 8; ++v) {
            float x = acc[v];
            x = x > 0.f ? x : 0.01f * x;           // leaky relu
            int rr = rbase + v;
            fHp[rr * 256 + nt * 16 + c] = x;
            sHp[rr * 256 + nt * 16 + c] = f2bf(x);
        }
    }
    __syncthreads();

    // GRU gates per 16-col H tile.  W_ih^T / W_hh^T are K=256, N=768 (48 ntiles).
    for (int j = 0; j < 16; ++j) {
        f32x8 axr = bcast8(b_ih[j * 16 + c]);
        f32x8 axz = bcast8(b_ih[256 + j * 16 + c]);
        f32x8 axn = bcast8(b_ih[512 + j * 16 + c]);
        f32x8 ahr = bcast8(b_hh[j * 16 + c]);
        f32x8 ahz = bcast8(b_hh[256 + j * 16 + c]);
        f32x8 ahn = bcast8(b_hh[512 + j * 16 + c]);
        for (int kt = 0; kt < 8; ++kt) {
            bf16x16 a1 = a_frag(sAgg, 256, kt * 32);
            bf16x16 a2 = a_frag(sHp, 256, kt * 32);
            axr = wmma_bf16(a1, b_frag(WihP + (kt * 48 + j) * 512), axr);
            axz = wmma_bf16(a1, b_frag(WihP + (kt * 48 + 16 + j) * 512), axz);
            axn = wmma_bf16(a1, b_frag(WihP + (kt * 48 + 32 + j) * 512), axn);
            ahr = wmma_bf16(a2, b_frag(WhhP + (kt * 48 + j) * 512), ahr);
            ahz = wmma_bf16(a2, b_frag(WhhP + (kt * 48 + 16 + j) * 512), ahz);
            ahn = wmma_bf16(a2, b_frag(WhhP + (kt * 48 + 32 + j) * 512), ahn);
        }
#pragma unroll
        for (int v = 0; v < 8; ++v) {
            int rr = rbase + v;
            float rg = sigm(axr[v] + ahr[v]);
            float zg = sigm(axz[v] + ahz[v]);
            float ng = tanhf(axn[v] + rg * ahn[v]);
            float hp = fHp[rr * 256 + j * 16 + c];
            float h = (1.f - zg) * ng + zg * hp;
            fHn[rr * 256 + j * 16 + c] = h;
            atomicAdd(&sSum[rr], h);
            atomicAdd(&sSq[rr], h * h);
        }
    }
    __syncthreads();

    // LayerNorm + write bf16 h_new
    for (int i = lane; i < 16 * 256; i += 32) {
        int r = i >> 8, cc = i & 255;
        float mu = sSum[r] * (1.f / 256.f);
        float var = sSq[r] * (1.f / 256.f) - mu * mu;
        float y = (fHn[i] - mu) * rsqrtf(var + 1e-5f) * ln_g[cc] + ln_b[cc];
        hnew_bf[(r0 + r) * 256 + cc] = f2bf(y);
    }
}

// ---------------------------------------------------------------------------
// Stage 4: candidate layer + rank score.  A=[te|hr|h_new[tail]] (16x512)
// @ cand_W (512x256), fused lrelu -> dot(rank_W[:256]) + qdot[batch] + b.
// ---------------------------------------------------------------------------
__global__ __launch_bounds__(32)
void cand_kernel(const float* __restrict__ entity_emb,
                 const float* __restrict__ relation_emb,
                 const unsigned short* __restrict__ hnew_bf,
                 const unsigned short* __restrict__ candP,
                 const float* __restrict__ cand_b,
                 const float* __restrict__ rank_W, const float* __restrict__ rank_b,
                 const float* __restrict__ qdot,
                 const int* __restrict__ tail_ent, const int* __restrict__ edge_rel,
                 const int* __restrict__ tail_node, const int* __restrict__ batch_idx,
                 float* __restrict__ out) {
    __shared__ unsigned short As[16 * 512];
    __shared__ float sS[16];
    __shared__ int sT[16], sR[16], sN[16], sB[16];
    const int lane = threadIdx.x;
    const int e0 = blockIdx.x * 16;
    if (lane < 16) {
        sT[lane] = tail_ent[e0 + lane];
        sR[lane] = edge_rel[e0 + lane];
        sN[lane] = tail_node[e0 + lane];
        sB[lane] = batch_idx[e0 + lane];
        sS[lane] = 0.f;
    }
    __syncthreads();
    for (int i = lane; i < 16 * 128; i += 32) { int r = i >> 7, c = i & 127; As[r * 512 + c]       = f2bf(entity_emb[sT[r] * 128 + c]); }
    for (int i = lane; i < 16 * 128; i += 32) { int r = i >> 7, c = i & 127; As[r * 512 + 128 + c] = f2bf(relation_emb[sR[r] * 128 + c]); }
    for (int i = lane; i < 16 * 256; i += 32) { int r = i >> 8, c = i & 255; As[r * 512 + 256 + c] = hnew_bf[sN[r] * 256 + c]; }
    __syncthreads();
    const int c = lane & 15;
    const int rbase = (lane < 16) ? 0 : 8;
    for (int nt = 0; nt < 16; ++nt) {
        f32x8 acc = bcast8(cand_b[nt * 16 + c]);
        for (int kt = 0; kt < 16; ++kt)
            acc = wmma_bf16(a_frag(As, 512, kt * 32), b_frag(candP + (kt * 16 + nt) * 512), acc);
        float rw = rank_W[nt * 16 + c];
#pragma unroll
        for (int v = 0; v < 8; ++v) {
            float x = acc[v];
            x = x > 0.f ? x : 0.01f * x;           // leaky relu
            atomicAdd(&sS[rbase + v], x * rw);
        }
    }
    __syncthreads();
    if (lane < 16) out[e0 + lane] = sS[lane] + qdot[sB[lane]] + rank_b[0];
}

// ---------------------------------------------------------------------------
// Launcher
// ---------------------------------------------------------------------------
extern "C" void kernel_launch(void* const* d_in, const int* in_sizes, int n_in,
                              void* d_out, int out_size, void* d_ws, size_t ws_size,
                              hipStream_t stream) {
    (void)in_sizes; (void)n_in; (void)out_size; (void)ws_size;
    const int M = 200000, N_NEW = 50000;

    const float* entity_emb    = (const float*)d_in[0];
    const float* relation_emb  = (const float*)d_in[1];
    const float* node_hidden   = (const float*)d_in[2];
    const float* query_repr    = (const float*)d_in[3];
    const float* Ws            = (const float*)d_in[4];
    const float* Wr            = (const float*)d_in[5];
    const float* Wqr           = (const float*)d_in[6];
    const float* bqr           = (const float*)d_in[7];
    const float* w_alpha       = (const float*)d_in[8];
    const float* b_alpha       = (const float*)d_in[9];
    const float* W_ih          = (const float*)d_in[10];
    const float* W_hh          = (const float*)d_in[11];
    const float* b_ih          = (const float*)d_in[12];
    const float* b_hh          = (const float*)d_in[13];
    const float* We2h_W        = (const float*)d_in[14];
    const float* We2h_b        = (const float*)d_in[15];
    const float* cand_W        = (const float*)d_in[16];
    const float* cand_b        = (const float*)d_in[17];
    const float* rank_W        = (const float*)d_in[18];
    const float* rank_b        = (const float*)d_in[19];
    const float* ln_g          = (const float*)d_in[20];
    const float* ln_b          = (const float*)d_in[21];
    const int* head_node       = (const int*)d_in[22];
    const int* edge_rel        = (const int*)d_in[23];
    const int* tail_ent        = (const int*)d_in[24];
    const int* tail_node       = (const int*)d_in[25];
    const int* query_rel       = (const int*)d_in[26];
    const int* batch_idx       = (const int*)d_in[27];
    const int* tail_node_ent   = (const int*)d_in[28];
    float* out = (float*)d_out;

    char* ws = (char*)d_ws;
    size_t off = 0;
    auto alloc = [&](size_t bytes) -> void* {
        void* p = ws + off;
        off = (off + bytes + 255) & ~(size_t)255;
        return p;
    };
    unsigned short* W1p   = (unsigned short*)alloc(512 * 256 * 2);
    unsigned short* candP = (unsigned short*)alloc(512 * 256 * 2);
    unsigned short* We2hP = (unsigned short*)alloc(128 * 256 * 2);
    unsigned short* WihP  = (unsigned short*)alloc(256 * 768 * 2);
    unsigned short* WhhP  = (unsigned short*)alloc(256 * 768 * 2);
    float* alpha          = (float*)alloc((size_t)M * 4);
    float* qdot           = (float*)alloc(256 * 4);
    float* agg            = (float*)alloc((size_t)N_NEW * 256 * 4);
    unsigned short* hnewP = (unsigned short*)alloc((size_t)N_NEW * 256 * 2);

    hipMemsetAsync(agg, 0, (size_t)N_NEW * 256 * 4, stream);

    pack_w1_kernel<<<131072 / 256, 256, 0, stream>>>(Ws, Wr, Wqr, W1p);
    pack_b_kernel<<<131072 / 256, 256, 0, stream>>>(cand_W, candP, 512, 256, 0);
    pack_b_kernel<<<32768 / 256, 256, 0, stream>>>(We2h_W, We2hP, 128, 256, 0);
    pack_b_kernel<<<196608 / 256, 256, 0, stream>>>(W_ih, WihP, 256, 768, 1);
    pack_b_kernel<<<196608 / 256, 256, 0, stream>>>(W_hh, WhhP, 256, 768, 1);
    qdot_kernel<<<256, 256, 0, stream>>>(query_repr, rank_W, qdot);

    attn_alpha_kernel<<<M / 16, 32, 0, stream>>>(node_hidden, relation_emb, W1p, bqr,
                                                 w_alpha, b_alpha, head_node, edge_rel,
                                                 query_rel, alpha);
    scatter_kernel<<<M, 256, 0, stream>>>(relation_emb, entity_emb, alpha,
                                          edge_rel, tail_ent, tail_node, agg);
    gru_kernel<<<N_NEW / 16, 32, 0, stream>>>(entity_emb, agg, We2hP, We2h_b,
                                              WihP, WhhP, b_ih, b_hh, ln_g, ln_b,
                                              tail_node_ent, hnewP);
    cand_kernel<<<M / 16, 32, 0, stream>>>(entity_emb, relation_emb, hnewP, candP,
                                           cand_b, rank_W, rank_b, qdot,
                                           tail_ent, edge_rel, tail_node, batch_idx, out);
}
